// Physics_Attention_Irregular_Mesh_5162550690714
// MI455X (gfx1250) — matrix operand order repaired and checked by reading, not compiled, not run on hardware
//
// MI455X / gfx1250 implementation of Physics_Attention_Irregular_Mesh.
//
// Pipeline (all on `stream`, deterministic, no atomics on floats):
//  1. x -> bf16 (L2-resident, 32 MB)
//  2. fold Wxp@Wq -> Wcomb ; repack weight matrices into WMMA B-fragment order
//  3. WMMA GEMM: fxT[b,h,c,n] = (x@Wfx+b)^T            (v_wmma_f32_16x16x32_bf16)
//  4. WMMA GEMM + fused softmax: w[b,h,n,g], wT[b,h,g,n]
//  5. denom[b,h,g] = sum_n w  (tree reduction)
//  6. WMMA: tok'[c,g] = fxT @ w   (K=16384, 8-way K-split, deterministic reduce)
//  7. Erwin ball-tree transformer: 1 workgroup / point set, all-LDS (≈54 KB)
//  8. WMMA: out[n,c] = w @ tok_out^T
//  9. WMMA GEMM: y = out @ Wout + b  -> d_out (fp32)
//
// Assumptions (compile-only environment): inputs flattened in setup_inputs()
// insertion order (x, pos, then params depth-first); ws_size >= ~245 MB.

#include <hip/hip_runtime.h>

typedef __bf16 bf16;
typedef bf16  v8bf  __attribute__((ext_vector_type(8)));
typedef bf16  v16bf __attribute__((ext_vector_type(16)));
typedef float v8f   __attribute__((ext_vector_type(8)));

#define NPTS 16384

// ROT = Rx(pi/4) @ Rz(pi/4) rows (pos @ ROT.T uses rows of ROT)
__constant__ float ROTC[9] = {
    0.5f, -0.5f, 0.70710678f,
    0.70710678f, 0.70710678f, 0.0f,
   -0.5f,  0.5f, 0.70710678f };

// ---------------------------------------------------------------- fragments
__device__ inline v16bf cat8(v8bf lo, v8bf hi) {
  return __builtin_shufflevector(lo, hi, 0,1,2,3,4,5,6,7,8,9,10,11,12,13,14,15);
}
// A fragment 16x32 (MxK) from row-major [rows][lda] (bf16). contiguous-k order (B uses the same):
// lane M = lane&15 ; elem i -> K = i + (lane>=16?16:0)
__device__ inline v16bf frag_a(const bf16* __restrict__ p, int lda, int lane) {
  const bf16* q = p + (size_t)(lane & 15) * lda + ((lane < 16) ? 0 : 16);
  v8bf lo = *(const v8bf*)q;
  v8bf hi = *(const v8bf*)(q + 8);
  return cat8(lo, hi);
}
// B fragment 32x16 (KxN) from *transposed* row-major source [N rows][ldb over K]:
// lane N = lane&15 ; elem i -> K = i + (lane>=16?16:0)
__device__ inline v16bf frag_bt(const bf16* __restrict__ p, int ldb, int lane) {
  const bf16* q = p + (size_t)(lane & 15) * ldb + ((lane < 16) ? 0 : 16);
  v8bf lo = *(const v8bf*)q;
  v8bf hi = *(const v8bf*)(q + 8);
  return cat8(lo, hi);
}
// B fragment from pre-packed weight tile: 32 lanes x 16 contiguous bf16
__device__ inline v16bf frag_bp(const bf16* __restrict__ tile, int lane) {
  return *(const v16bf*)(tile + lane * 16);
}
__device__ inline v8f wmma_bf16(v16bf a, v16bf b, v8f c) {
  return __builtin_amdgcn_wmma_f32_16x16x32_bf16(false, a, false, b, (short)0, c,
                                                 false, false);
}

// ---------------------------------------------------------------- utilities
__global__ void k_f32bf16(const float* __restrict__ in, bf16* __restrict__ out,
                          size_t n) {
  size_t i = (size_t)blockIdx.x * blockDim.x + threadIdx.x;
  size_t st = (size_t)gridDim.x * blockDim.x;
  for (; i < n; i += st) out[i] = (bf16)in[i];
}

// Repack fp32 weight W[K][Nc] row-major into B-fragment tiles:
// Bp[((nt*(K/32)+kt)*32 + lane)*16 + e] = W[kt*32 + e + (lane<16?0:16)][nt*16 + (lane&15)]
__global__ void k_repack(const float* __restrict__ W, bf16* __restrict__ Bp,
                         int K, int Nc) {
  size_t total = (size_t)K * Nc;
  int ktiles = K >> 5;
  for (size_t i = (size_t)blockIdx.x * 256 + threadIdx.x; i < total;
       i += (size_t)gridDim.x * 256) {
    int e = (int)(i & 15);
    int lane = (int)((i >> 4) & 31);
    size_t t = i >> 9;
    int kt = (int)(t % ktiles);
    int nt = (int)(t / ktiles);
    int kk = kt * 32 + e + ((lane < 16) ? 0 : 16);
    int nn = nt * 16 + (lane & 15);
    Bp[i] = (bf16)W[(size_t)kk * Nc + nn];
  }
}

// Wcomb[k][h*64+g] = sum_d WxpW[k][h*64+d] * Wq[d][g]
__global__ void k_wcomb(const float* __restrict__ WxpW,
                        const float* __restrict__ Wq, float* __restrict__ Wc) {
  int i = blockIdx.x * 256 + threadIdx.x;
  if (i >= 256 * 512) return;
  int k = i >> 9, j = i & 511, h = j >> 6, g = j & 63;
  float a = 0.f;
  for (int d = 0; d < 64; ++d) a += WxpW[k * 512 + h * 64 + d] * Wq[d * 64 + g];
  Wc[i] = a;
}
__global__ void k_bcomb(const float* __restrict__ xpB, const float* __restrict__ Wq,
                        const float* __restrict__ slB, float* __restrict__ bc) {
  int j = blockIdx.x * 256 + threadIdx.x;
  if (j >= 512) return;
  int h = j >> 6, g = j & 63;
  float a = slB[g];
  for (int d = 0; d < 64; ++d) a += xpB[h * 64 + d] * Wq[d * 64 + g];
  bc[j] = a;
}

// ---------------------------------------------------------------- big GEMM
// BM=128, BN=64, BK=32; 8 waves, each 32x32 (2x2 WMMA). MODE 0: fp32 row-major
// out (+bias). MODE 1: bf16 out in fxT layout [b][h][c][NPTS] (+bias).
template <int MODE>
__global__ __launch_bounds__(256) void k_gemm(const bf16* __restrict__ A,
                                              const bf16* __restrict__ Bp,
                                              const float* __restrict__ bias,
                                              float* __restrict__ outF,
                                              bf16* __restrict__ outH,
                                              int M, int K, int Nc) {
  const int lane = threadIdx.x & 31, w = threadIdx.x >> 5;
  const int wm = w & 3, wn = w >> 2;
  const int m0 = blockIdx.x * 128 + wm * 32;
  const int nBase = blockIdx.y * 64 + wn * 32;
  const int ktiles = K >> 5;
  v8f acc[2][2] = {};
  const bf16* Arow = A + (size_t)m0 * K;
  const bf16* Bt0 = Bp + (size_t)((nBase >> 4) + 0) * ktiles * 512;
  const bf16* Bt1 = Bp + (size_t)((nBase >> 4) + 1) * ktiles * 512;
  for (int kt = 0; kt < ktiles; ++kt) {
    const int k0 = kt << 5;
    if (kt + 1 < ktiles) __builtin_prefetch(Arow + k0 + 32, 0, 1);
    v16bf a0 = frag_a(Arow + k0, K, lane);
    v16bf a1 = frag_a(Arow + (size_t)16 * K + k0, K, lane);
    v16bf b0 = frag_bp(Bt0 + (size_t)kt * 512, lane);
    v16bf b1 = frag_bp(Bt1 + (size_t)kt * 512, lane);
    acc[0][0] = wmma_bf16(a0, b0, acc[0][0]);
    acc[0][1] = wmma_bf16(a0, b1, acc[0][1]);
    acc[1][0] = wmma_bf16(a1, b0, acc[1][0]);
    acc[1][1] = wmma_bf16(a1, b1, acc[1][1]);
  }
#pragma unroll
  for (int rm = 0; rm < 2; ++rm)
#pragma unroll
    for (int rn = 0; rn < 2; ++rn) {
      int nCol = nBase + rn * 16 + (lane & 15);
      float bv = bias ? bias[nCol] : 0.f;
#pragma unroll
      for (int r = 0; r < 8; ++r) {
        int mRow = m0 + rm * 16 + r + ((lane < 16) ? 0 : 8);
        float v = acc[rm][rn][r] + bv;
        if constexpr (MODE == 0) {
          outF[(size_t)mRow * Nc + nCol] = v;
        } else {
          int b = mRow >> 14, npt = mRow & (NPTS - 1);
          int h = nCol >> 6, cc = nCol & 63;
          outH[(((size_t)(b * 8 + h) * 64 + cc) << 14) + npt] = (bf16)v;
        }
      }
    }
}

// ---------------------------------------------------- fused logits + softmax
// Block = 64 rows x one head (64 g). w and wT written in bf16.
__global__ __launch_bounds__(256) void k_slice(const bf16* __restrict__ A,
                                               const bf16* __restrict__ Bp,
                                               const float* __restrict__ bcomb,
                                               const float* __restrict__ temp,
                                               bf16* __restrict__ wN,
                                               bf16* __restrict__ wT) {
  __shared__ float Z[64 * 64];
  const int lane = threadIdx.x & 31, w = threadIdx.x >> 5;
  const int wm = w & 3, wn = w >> 2;
  const int h = blockIdx.y;
  const int m0 = blockIdx.x * 64;
  const int K = 256, ktiles = 8;
  v8f acc[2] = {};
  const bf16* Arow = A + (size_t)(m0 + wm * 16) * K;
  const int ntBase = (h * 64 + wn * 32) >> 4;
  for (int kt = 0; kt < ktiles; ++kt) {
    v16bf a0 = frag_a(Arow + kt * 32, K, lane);
    v16bf b0 = frag_bp(Bp + ((size_t)(ntBase + 0) * ktiles + kt) * 512, lane);
    v16bf b1 = frag_bp(Bp + ((size_t)(ntBase + 1) * ktiles + kt) * 512, lane);
    acc[0] = wmma_bf16(a0, b0, acc[0]);
    acc[1] = wmma_bf16(a0, b1, acc[1]);
  }
#pragma unroll
  for (int rn = 0; rn < 2; ++rn) {
    int g = wn * 32 + rn * 16 + (lane & 15);
    float bv = bcomb[h * 64 + g];
#pragma unroll
    for (int r = 0; r < 8; ++r) {
      int rl = wm * 16 + r + ((lane < 16) ? 0 : 8);
      Z[rl * 64 + g] = acc[rn][r] + bv;
    }
  }
  __syncthreads();
  const float invT = 1.f / temp[h];
  for (int r = threadIdx.x; r < 64; r += 256) {
    float mx = -1e30f;
    for (int g = 0; g < 64; ++g) mx = fmaxf(mx, Z[r * 64 + g]);
    float s = 0.f;
    for (int g = 0; g < 64; ++g) {
      float e = __expf((Z[r * 64 + g] - mx) * invT);
      Z[r * 64 + g] = e;
      s += e;
    }
    float rinv = 1.f / s;
    int mRow = m0 + r;
    int b = mRow >> 14, npt = mRow & (NPTS - 1);
    size_t bh = (size_t)b * 8 + h;
    for (int g = 0; g < 64; ++g) {
      bf16 v = (bf16)(Z[r * 64 + g] * rinv);
      wN[((bh << 14) + npt) * 64 + g] = v;
      wT[(bh * 64 + g) * NPTS + npt] = v;
    }
  }
}

// denom[bh*64+g] = sum_n wT row
__global__ __launch_bounds__(256) void k_denom(const bf16* __restrict__ wT,
                                               float* __restrict__ denom) {
  __shared__ float red[256];
  const bf16* row = wT + (size_t)blockIdx.x * NPTS;
  float s = 0.f;
  for (int i = threadIdx.x; i < NPTS; i += 256) s += (float)row[i];
  red[threadIdx.x] = s;
  __syncthreads();
  for (int st = 128; st > 0; st >>= 1) {
    if (threadIdx.x < st) red[threadIdx.x] += red[threadIdx.x + st];
    __syncthreads();
  }
  if (threadIdx.x == 0) denom[blockIdx.x] = red[0];
}

// tok partials: T'[c,g] = sum_n fxT[bh][c][n] * wT[bh][g][n], K-split over n.
__global__ __launch_bounds__(256) void k_tok(const bf16* __restrict__ fxT,
                                             const bf16* __restrict__ wT,
                                             float* __restrict__ part) {
  const int bh = blockIdx.x, ks = blockIdx.y;
  const int lane = threadIdx.x & 31, w = threadIdx.x >> 5;
  const int wm = w & 3, wn = w >> 2;
  const bf16* Abase = fxT + ((size_t)bh * 64 + wm * 16) * NPTS;
  const bf16* Bbase = wT + (size_t)bh * 64 * NPTS;
  v8f acc[2] = {};
  const int kStart = ks * 2048;
  for (int kt = 0; kt < 64; ++kt) {
    const int k0 = kStart + kt * 32;
    if (kt + 1 < 64) __builtin_prefetch(Abase + k0 + 32, 0, 1);
    v16bf a = frag_a(Abase + k0, NPTS, lane);
    v16bf b0 = frag_bt(Bbase + (size_t)(wn * 32) * NPTS + k0, NPTS, lane);
    v16bf b1 = frag_bt(Bbase + (size_t)(wn * 32 + 16) * NPTS + k0, NPTS, lane);
    acc[0] = wmma_bf16(a, b0, acc[0]);
    acc[1] = wmma_bf16(a, b1, acc[1]);
  }
  float* P = part + ((size_t)ks * 32 + bh) * 4096;
#pragma unroll
  for (int rn = 0; rn < 2; ++rn)
#pragma unroll
    for (int r = 0; r < 8; ++r) {
      int cRow = wm * 16 + r + ((lane < 16) ? 0 : 8);
      int g = wn * 32 + rn * 16 + (lane & 15);
      P[cRow * 64 + g] = acc[rn][r];
    }
}

// deterministic reduce of K-split partials + divide by denom -> tok[bh][g][c]
__global__ void k_tokred(const float* __restrict__ part,
                         const float* __restrict__ denom,
                         float* __restrict__ tok) {
  int i = blockIdx.x * 256 + threadIdx.x;
  if (i >= 32 * 4096) return;
  int bh = i >> 12, rem = i & 4095, cRow = rem >> 6, g = rem & 63;
  float s = 0.f;
  for (int ks = 0; ks < 8; ++ks) s += part[((size_t)ks * 32 + bh) * 4096 + rem];
  tok[((size_t)bh * 64 + g) * 64 + cRow] = s / (denom[bh * 64 + g] + 1e-5f);
}

// out[n, h*64+c] = sum_g w[bh][n][g] * OT[bh][c][g]
__global__ __launch_bounds__(256) void k_outmix(const bf16* __restrict__ wN,
                                                const bf16* __restrict__ OT,
                                                bf16* __restrict__ outbuf) {
  const int bh = blockIdx.y, nt = blockIdx.x;
  const int lane = threadIdx.x & 31, w = threadIdx.x >> 5;
  const int wm = w & 3, wn = w >> 2;
  const int n0 = nt * 128 + wm * 32;
  const bf16* Abase = wN + ((size_t)bh * NPTS + n0) * 64;
  const bf16* Bbase = OT + (size_t)bh * 4096;
  v8f acc[2][2] = {};
  for (int kt = 0; kt < 2; ++kt) {
    const int k0 = kt * 32;
    v16bf a0 = frag_a(Abase + k0, 64, lane);
    v16bf a1 = frag_a(Abase + 16 * 64 + k0, 64, lane);
    v16bf b0 = frag_bt(Bbase + (wn * 32) * 64 + k0, 64, lane);
    v16bf b1 = frag_bt(Bbase + (wn * 32 + 16) * 64 + k0, 64, lane);
    acc[0][0] = wmma_bf16(a0, b0, acc[0][0]);
    acc[0][1] = wmma_bf16(a0, b1, acc[0][1]);
    acc[1][0] = wmma_bf16(a1, b0, acc[1][0]);
    acc[1][1] = wmma_bf16(a1, b1, acc[1][1]);
  }
  const int b = bh >> 3, h = bh & 7;
#pragma unroll
  for (int rm = 0; rm < 2; ++rm)
#pragma unroll
    for (int rn = 0; rn < 2; ++rn)
#pragma unroll
      for (int r = 0; r < 8; ++r) {
        int n = n0 + rm * 16 + r + ((lane < 16) ? 0 : 8);
        int cc = wn * 32 + rn * 16 + (lane & 15);
        outbuf[((size_t)b * NPTS + n) * 512 + h * 64 + cc] = (bf16)acc[rm][rn][r];
      }
}

// ================================================================== Erwin
struct BlockP {
  const float *norm1, *norm2, *peW, *peB, *qkvW, *qkvB, *projW, *projB,
      *sigma, *w1W, *w1B, *w2W, *w2B, *w3W, *w3B;
};
struct ErwinParams {
  const float *embW, *embB, *poolW, *poolB, *poolN, *unpW, *unpB, *unpN;
  BlockP bp[6];  // enc0[0..1], enc1[0..1], dec0[0..1]
};

__device__ void rms_rows(const float* __restrict__ in, float* __restrict__ out,
                         const float* __restrict__ wn, int n, int c, float* RS,
                         int tid) {
  for (int r = tid; r < n; r += 256) {
    float ss = 0.f;
    for (int t = 0; t < c; ++t) { float v = in[r * c + t]; ss += v * v; }
    RS[r] = rsqrtf(ss / (float)c + 1e-5f);
  }
  __syncthreads();
  for (int i = tid; i < n * c; i += 256) out[i] = in[i] * RS[i / c] * wn[i % c];
  __syncthreads();
}

// stable median-split ball-tree ordering (matches jnp.argsort stable sort)
__device__ void build_perm(const float* __restrict__ p3, int n, int* perm,
                           int tid) {
  if (tid < n) perm[tid] = tid;
  __syncthreads();
  for (int s = n; s > 1; s >>= 1) {
    int g = n / s;
    if (tid < g) {
      int base = tid * s;
      float mn0 = 1e30f, mn1 = 1e30f, mn2 = 1e30f;
      float mx0 = -1e30f, mx1 = -1e30f, mx2 = -1e30f;
      for (int i = 0; i < s; ++i) {
        int q = perm[base + i];
        float a = p3[q * 3], b = p3[q * 3 + 1], d = p3[q * 3 + 2];
        mn0 = fminf(mn0, a); mx0 = fmaxf(mx0, a);
        mn1 = fminf(mn1, b); mx1 = fmaxf(mx1, b);
        mn2 = fminf(mn2, d); mx2 = fmaxf(mx2, d);
      }
      int ax = 0;
      float best = mx0 - mn0;
      if (mx1 - mn1 > best) { ax = 1; best = mx1 - mn1; }
      if (mx2 - mn2 > best) { ax = 2; }
      for (int i = 1; i < s; ++i) {  // stable insertion sort
        int pv = perm[base + i];
        float kv = p3[pv * 3 + ax];
        int j = i - 1;
        while (j >= 0 && p3[perm[base + j] * 3 + ax] > kv) {
          perm[base + j + 1] = perm[base + j];
          --j;
        }
        perm[base + j + 1] = pv;
      }
    }
    __syncthreads();
  }
}

__device__ void ball_block(float* X, float* H, float* QKV, float* ATT, float* OB,
                           float* RS, float* BC, const float* posb, int n, int c,
                           int m, int heads, const BlockP& bp, int tid) {
  rms_rows(X, H, bp.norm1, n, c, RS, tid);
  if (tid < 6) {  // 2 balls x 3 coords
    int b2 = tid / 3, a = tid % 3;
    float s = 0.f;
    for (int i = 0; i < m; ++i) s += posb[(b2 * m + i) * 3 + a];
    BC[tid] = s / (float)m;
  }
  __syncthreads();
  for (int i = tid; i < n * c; i += 256) {  // h += pe(rel)
    int r = i / c, t = i % c, b2 = r / m;
    float r0 = posb[r * 3 + 0] - BC[b2 * 3 + 0];
    float r1 = posb[r * 3 + 1] - BC[b2 * 3 + 1];
    float r2 = posb[r * 3 + 2] - BC[b2 * 3 + 2];
    H[i] += bp.peB[t] + r0 * bp.peW[t] + r1 * bp.peW[c + t] + r2 * bp.peW[2 * c + t];
  }
  __syncthreads();
  const int hd = c / heads;
  const float scale = rsqrtf((float)hd);
  for (int b2 = 0; b2 < 2; ++b2) {
    const float* hb = H + b2 * m * c;
    const float* pb = posb + b2 * m * 3;
    for (int hh = 0; hh < heads; ++hh) {
      for (int i = tid; i < 3 * m * hd; i += 256) {  // q,k,v slices for head hh
        int which = i / (m * hd), rr = (i / hd) % m, d = i % hd;
        int col = which * c + hh * hd + d;
        float a = bp.qkvB[col];
        const float* hr = hb + rr * c;
        for (int t = 0; t < c; ++t) a += hr[t] * bp.qkvW[t * 3 * c + col];
        QKV[i] = a;
      }
      __syncthreads();
      const float *Qh = QKV, *Kh = QKV + m * hd, *Vh = QKV + 2 * m * hd;
      const float sg = bp.sigma[hh];
      for (int i = tid; i < m * m; i += 256) {
        int r = i / m, j = i % m;
        float dp = 0.f;
        for (int d = 0; d < hd; ++d) dp += Qh[r * hd + d] * Kh[j * hd + d];
        float dx = pb[r * 3] - pb[j * 3], dy = pb[r * 3 + 1] - pb[j * 3 + 1],
              dz = pb[r * 3 + 2] - pb[j * 3 + 2];
        ATT[i] = dp * scale + sg * sqrtf(dx * dx + dy * dy + dz * dz + 1e-12f);
      }
      __syncthreads();
      for (int r = tid; r < m; r += 256) {  // softmax rows
        float mx = -1e30f;
        for (int j = 0; j < m; ++j) mx = fmaxf(mx, ATT[r * m + j]);
        float s = 0.f;
        for (int j = 0; j < m; ++j) {
          float e = __expf(ATT[r * m + j] - mx);
          ATT[r * m + j] = e;
          s += e;
        }
        float iv = 1.f / s;
        for (int j = 0; j < m; ++j) ATT[r * m + j] *= iv;
      }
      __syncthreads();
      for (int i = tid; i < m * hd; i += 256) {
        int r = i / hd, d = i % hd;
        float a = 0.f;
        for (int j = 0; j < m; ++j) a += ATT[r * m + j] * Vh[j * hd + d];
        OB[r * c + hh * hd + d] = a;
      }
      __syncthreads();
    }
    for (int i = tid; i < m * c; i += 256) {  // proj + residual
      int r = i / c, t = i % c;
      float a = bp.projB[t];
      const float* orow = OB + r * c;
      for (int u = 0; u < c; ++u) a += orow[u] * bp.projW[u * c + t];
      X[(b2 * m + r) * c + t] += a;
    }
    __syncthreads();
  }
  // gated MLP, hidden chunked by 32 (OB reused as chunk buffer)
  rms_rows(X, H, bp.norm2, n, c, RS, tid);
  const int hid = 4 * c;
  for (int k0 = 0; k0 < hid; k0 += 32) {
    for (int i = tid; i < n * 32; i += 256) {
      int r = i >> 5, kk = k0 + (i & 31);
      float a1 = bp.w1B[kk], a2 = bp.w2B[kk];
      const float* hr = H + r * c;
      for (int t = 0; t < c; ++t) {
        float hv = hr[t];
        a1 += hv * bp.w1W[t * hid + kk];
        a2 += hv * bp.w2W[t * hid + kk];
      }
      OB[i] = a2 * (a1 / (1.f + __expf(-a1)));  // w2(h) * silu(w1(h))
    }
    __syncthreads();
    for (int i = tid; i < n * c; i += 256) {
      int r = i / c, t = i % c;
      float a = 0.f;
      for (int kk = 0; kk < 32; ++kk) a += OB[(r << 5) + kk] * bp.w3W[(k0 + kk) * c + t];
      X[i] += a;
    }
    __syncthreads();
  }
  for (int i = tid; i < n * c; i += 256) X[i] += bp.w3B[i % c];
  __syncthreads();
}

__device__ void run_pair(float* X, float* H, float* QKV, float* ATT, float* OB,
                         float* RS, float* BC, const float* posb, float* rposbuf,
                         const int* rot, const int* rinv, int n, int c, int m,
                         int heads, const BlockP& b0, const BlockP& b1, int tid) {
  ball_block(X, H, QKV, ATT, OB, RS, BC, posb, n, c, m, heads, b0, tid);
  for (int i = tid; i < n * c; i += 256) H[i] = X[rot[i / c] * c + (i % c)];
  for (int i = tid; i < n * 3; i += 256) rposbuf[i] = posb[rot[i / 3] * 3 + (i % 3)];
  __syncthreads();
  for (int i = tid; i < n * c; i += 256) X[i] = H[i];
  __syncthreads();
  ball_block(X, H, QKV, ATT, OB, RS, BC, rposbuf, n, c, m, heads, b1, tid);
  for (int i = tid; i < n * c; i += 256) H[i] = X[rinv[i / c] * c + (i % c)];
  __syncthreads();
  for (int i = tid; i < n * c; i += 256) X[i] = H[i];
  __syncthreads();
}

#define ERWIN_LDS_BYTES ((13544 + 320) * 4)

__global__ __launch_bounds__(256) void k_erwin(const float* __restrict__ tok,
                                               const float* __restrict__ posg,
                                               float* __restrict__ skipx,
                                               bf16* __restrict__ OT,
                                               ErwinParams P) {
  extern __shared__ float sm[];
  float* X = sm;                 // 4096
  float* H = X + 4096;           // 4096
  float* QKV = H + 4096;         // 1536
  float* ATT = QKV + 1536;       // 1024
  float* OB = ATT + 1024;        // 2048
  float* POS = OB + 2048;        // 192
  float* RPOS = POS + 192;       // 192
  float* CTR = RPOS + 192;       // 96
  float* SKPP = CTR + 96;        // 192
  float* RS = SKPP + 192;        // 64
  float* BC = RS + 64;           // 8
  int* PERM = (int*)(BC + 8);    // 64
  int* INV = PERM + 64;          // 64
  int* R64 = INV + 64;           // 64
  int* RI64 = R64 + 64;          // 64
  int* R32 = RI64 + 64;          // 32
  int* RI32 = R32 + 32;          // 32
  const int S = blockIdx.x, tid = threadIdx.x;

  for (int i = tid; i < 192; i += 256) RPOS[i] = posg[(size_t)S * 192 + i];
  __syncthreads();
  build_perm(RPOS, 64, PERM, tid);
  if (tid < 64) INV[PERM[tid]] = tid;
  __syncthreads();
  for (int i = tid; i < 64; i += 256) {
    int sr = PERM[i];
    POS[i * 3 + 0] = RPOS[sr * 3 + 0];
    POS[i * 3 + 1] = RPOS[sr * 3 + 1];
    POS[i * 3 + 2] = RPOS[sr * 3 + 2];
  }
  __syncthreads();
  for (int i = tid; i < 4096; i += 256) {  // X = embed(tok)[perm]
    int r = i >> 6, ch = i & 63;
    const float* trow = tok + (size_t)S * 4096 + PERM[r] * 64;
    float a = P.embB[ch];
    for (int t = 0; t < 64; ++t) a += trow[t] * P.embW[t * 64 + ch];
    X[i] = a;
  }
  __syncthreads();
  for (int i = tid; i < 64; i += 256)  // rotated keys for layer perm
    for (int a = 0; a < 3; ++a)
      RPOS[i * 3 + a] = POS[i * 3] * ROTC[a * 3] + POS[i * 3 + 1] * ROTC[a * 3 + 1] +
                        POS[i * 3 + 2] * ROTC[a * 3 + 2];
  __syncthreads();
  build_perm(RPOS, 64, R64, tid);
  if (tid < 64) RI64[R64[tid]] = tid;
  __syncthreads();
  // enc0
  run_pair(X, H, QKV, ATT, OB, RS, BC, POS, RPOS, R64, RI64, 64, 64, 32, 4,
           P.bp[0], P.bp[1], tid);
  for (int i = tid; i < 4096; i += 256) skipx[(size_t)S * 4096 + i] = X[i];
  for (int i = tid; i < 192; i += 256) SKPP[i] = POS[i];
  __syncthreads();
  // pool (stride-2)
  if (tid < 96) {
    int j = tid / 3, a = tid % 3;
    CTR[tid] = 0.5f * (POS[(2 * j) * 3 + a] + POS[(2 * j + 1) * 3 + a]);
  }
  __syncthreads();
  for (int i = tid; i < 32 * 128; i += 256) {
    int j = i >> 7, o = i & 127;
    float a = P.poolB[o];
    const float* x0 = X + (2 * j) * 64;
    const float* x1 = X + (2 * j + 1) * 64;
    for (int t = 0; t < 64; ++t) a += x0[t] * P.poolW[t * 128 + o];
    for (int t = 0; t < 64; ++t) a += x1[t] * P.poolW[(64 + t) * 128 + o];
    for (int d = 0; d < 3; ++d) {
      a += (POS[(2 * j) * 3 + d] - CTR[j * 3 + d]) * P.poolW[(128 + d) * 128 + o];
      a += (POS[(2 * j + 1) * 3 + d] - CTR[j * 3 + d]) * P.poolW[(131 + d) * 128 + o];
    }
    H[i] = a;
  }
  __syncthreads();
  rms_rows(H, X, P.poolN, 32, 128, RS, tid);
  for (int i = tid; i < 32; i += 256)
    for (int a = 0; a < 3; ++a)
      RPOS[i * 3 + a] = CTR[i * 3] * ROTC[a * 3] + CTR[i * 3 + 1] * ROTC[a * 3 + 1] +
                        CTR[i * 3 + 2] * ROTC[a * 3 + 2];
  __syncthreads();
  build_perm(RPOS, 32, R32, tid);
  if (tid < 32) RI32[R32[tid]] = tid;
  __syncthreads();
  // enc1
  run_pair(X, H, QKV, ATT, OB, RS, BC, CTR, RPOS, R32, RI32, 32, 128, 16, 8,
           P.bp[2], P.bp[3], tid);
  // unpool + skip + rms
  for (int i = tid; i < 4096; i += 256) {
    int r = i >> 6, ch = i & 63, j = r >> 1, half = r & 1, o = half * 64 + ch;
    float a = P.unpB[o];
    const float* xr = X + j * 128;
    for (int t = 0; t < 128; ++t) a += xr[t] * P.unpW[t * 128 + o];
    for (int d = 0; d < 3; ++d) {
      a += (SKPP[(2 * j) * 3 + d] - CTR[j * 3 + d]) * P.unpW[(128 + d) * 128 + o];
      a += (SKPP[(2 * j + 1) * 3 + d] - CTR[j * 3 + d]) * P.unpW[(131 + d) * 128 + o];
    }
    H[i] = a + skipx[(size_t)S * 4096 + i];
  }
  __syncthreads();
  rms_rows(H, X, P.unpN, 64, 64, RS, tid);
  // dec0 (same rot as enc0: skip_pos == enc0 pos)
  run_pair(X, H, QKV, ATT, OB, RS, BC, SKPP, RPOS, R64, RI64, 64, 64, 32, 4,
           P.bp[4], P.bp[5], tid);
  // final inverse-perm, write OT[S][c][g] bf16 (B-fragment friendly for outmix)
  for (int i = tid; i < 4096; i += 256) {
    int ch = i >> 6, g = i & 63;
    OT[(size_t)S * 4096 + i] = (bf16)X[INV[g] * 64 + ch];
  }
}

// ================================================================== launch
static void fill_block(void* const* d_in, int base, BlockP& b) {
  b.norm1 = (const float*)d_in[base + 0];
  b.norm2 = (const float*)d_in[base + 1];
  b.peW   = (const float*)d_in[base + 2];
  b.peB   = (const float*)d_in[base + 3];
  b.qkvW  = (const float*)d_in[base + 4];
  b.qkvB  = (const float*)d_in[base + 5];
  b.projW = (const float*)d_in[base + 6];
  b.projB = (const float*)d_in[base + 7];
  b.sigma = (const float*)d_in[base + 8];
  b.w1W   = (const float*)d_in[base + 9];
  b.w1B   = (const float*)d_in[base + 10];
  b.w2W   = (const float*)d_in[base + 11];
  b.w2B   = (const float*)d_in[base + 12];
  b.w3W   = (const float*)d_in[base + 13];
  b.w3B   = (const float*)d_in[base + 14];
}

extern "C" void kernel_launch(void* const* d_in, const int* in_sizes, int n_in,
                              void* d_out, int out_size, void* d_ws, size_t ws_size,
                              hipStream_t stream) {
  (void)in_sizes; (void)n_in; (void)out_size; (void)ws_size;
  const float* x    = (const float*)d_in[0];
  const float* pos  = (const float*)d_in[1];
  const float* fxW  = (const float*)d_in[2];
  const float* fxB  = (const float*)d_in[3];
  const float* xpW  = (const float*)d_in[4];
  const float* xpB  = (const float*)d_in[5];
  const float* slW  = (const float*)d_in[6];
  const float* slB  = (const float*)d_in[7];
  const float* temp = (const float*)d_in[8];
  const float* outW = (const float*)d_in[9];
  const float* outB = (const float*)d_in[10];

  ErwinParams P;
  P.embW = (const float*)d_in[11];
  P.embB = (const float*)d_in[12];
  fill_block(d_in, 13, P.bp[0]);
  fill_block(d_in, 28, P.bp[1]);
  P.poolW = (const float*)d_in[43];
  P.poolB = (const float*)d_in[44];
  P.poolN = (const float*)d_in[45];
  fill_block(d_in, 46, P.bp[2]);
  fill_block(d_in, 61, P.bp[3]);
  P.unpW = (const float*)d_in[76];
  P.unpB = (const float*)d_in[77];
  P.unpN = (const float*)d_in[78];
  fill_block(d_in, 79, P.bp[4]);
  fill_block(d_in, 94, P.bp[5]);

  char* ws = (char*)d_ws;
  size_t off = 0;
  auto carve = [&](size_t bytes) -> char* {
    char* p = ws + off;
    off = (off + bytes + 255) & ~(size_t)255;
    return p;
  };
  bf16*  xb    = (bf16*)carve((size_t)65536 * 256 * 2);     // 32 MB (L2-resident)
  bf16*  fxT   = (bf16*)carve((size_t)32 * 64 * NPTS * 2);  // 64 MB
  bf16*  wN    = (bf16*)carve((size_t)32 * NPTS * 64 * 2);  // 64 MB
  bf16*  wT    = (bf16*)carve((size_t)32 * 64 * NPTS * 2);  // 64 MB
  bf16*  WfxP  = (bf16*)carve(262144);
  bf16*  WcmP  = (bf16*)carve(262144);
  bf16*  WoutP = (bf16*)carve(262144);
  float* Wcomb = (float*)carve(524288);
  float* bcomb = (float*)carve(2048);
  float* denom = (float*)carve(8192);
  float* part  = (float*)carve((size_t)8 * 32 * 4096 * 4);  // 4 MB
  float* tokb  = (float*)carve(524288);
  bf16*  OT    = (bf16*)carve(262144);
  float* skipx = (float*)carve(524288);
  bf16*  outbuf = fxT;  // alias: fxT dead after k_tok, outbuf written after

  k_f32bf16<<<dim3(8192), dim3(256), 0, stream>>>(x, xb, (size_t)65536 * 256);
  k_wcomb<<<dim3(512), dim3(256), 0, stream>>>(xpW, slW, Wcomb);
  k_bcomb<<<dim3(2), dim3(256), 0, stream>>>(xpB, slW, slB, bcomb);
  k_repack<<<dim3(512), dim3(256), 0, stream>>>(fxW, WfxP, 256, 512);
  k_repack<<<dim3(512), dim3(256), 0, stream>>>(Wcomb, WcmP, 256, 512);
  k_repack<<<dim3(512), dim3(256), 0, stream>>>(outW, WoutP, 512, 256);
  k_gemm<1><<<dim3(512, 8), dim3(256), 0, stream>>>(xb, WfxP, fxB, nullptr, fxT,
                                                    65536, 256, 512);
  k_slice<<<dim3(1024, 8), dim3(256), 0, stream>>>(xb, WcmP, bcomb, temp, wN, wT);
  k_denom<<<dim3(2048), dim3(256), 0, stream>>>(wT, denom);
  k_tok<<<dim3(32, 8), dim3(256), 0, stream>>>(fxT, wT, part);
  k_tokred<<<dim3(512), dim3(256), 0, stream>>>(part, denom, tokb);
  k_erwin<<<dim3(32), dim3(256), ERWIN_LDS_BYTES, stream>>>(tokb, pos, skipx, OT, P);
  k_outmix<<<dim3(128, 32), dim3(256), 0, stream>>>(wN, OT, outbuf);
  k_gemm<0><<<dim3(512, 4), dim3(256), 0, stream>>>(outbuf, WoutP, outB,
                                                    (float*)d_out, nullptr,
                                                    65536, 512, 256);
}
